// Fusion_39805756899995
// MI455X (gfx1250) — compile-verified, matrix operand order checked by reading, not run on hardware
//
#include <hip/hip_runtime.h>

// ---------------------------------------------------------------------------
// CDNA5 (gfx1250) fused decoder pipeline, wave32.
// GEMM work on v_wmma_f32_16x16x32_bf16; fragments loaded as b128 vectors
// (K-innermost layouts match the ISA 7.12.2 per-lane striping).
// ---------------------------------------------------------------------------

typedef __attribute__((ext_vector_type(16))) __bf16          v16bf;
typedef __attribute__((ext_vector_type(16))) unsigned short  v16u;
typedef __attribute__((ext_vector_type(8)))  float           v8f;
typedef __attribute__((ext_vector_type(4)))  unsigned int    u32x4;

__device__ inline unsigned short f32_to_bf16(float f) {
    union { float f; unsigned u; } x; x.f = f;
    unsigned r = x.u + 0x7FFFu + ((x.u >> 16) & 1u);   // round-to-nearest-even
    return (unsigned short)(r >> 16);
}
__device__ inline float bf16_to_f32(unsigned short h) {
    union { unsigned u; float f; } x; x.u = ((unsigned)h) << 16;
    return x.f;
}

struct FragPair { u32x4 lo, hi; };
__device__ inline v16u make_frag(u32x4 lo, u32x4 hi) {
    FragPair p{lo, hi};
    return __builtin_bit_cast(v16u, p);
}

// D(16x16 f32) = A(16x32 bf16) x B(32x16 bf16) + C
__device__ inline v8f wmma_bf16(const v16u& au, const v16u& bu, v8f c) {
    v16bf a = __builtin_bit_cast(v16bf, au);
    v16bf b = __builtin_bit_cast(v16bf, bu);
    return __builtin_amdgcn_wmma_f32_16x16x32_bf16(false, a, false, b,
                                                   (short)0, c, false, false);
}

#define HW 16384   // 128*128
#define NB 4
#define NC 128

// ---------------------------------------------------------------------------
// generic f32 -> bf16 conversion (used for [cout][cin] 1x1 weights etc.)
__global__ void k_cvt_bf16(const float* __restrict__ in,
                           unsigned short* __restrict__ out, int n) {
    int i = blockIdx.x * 256 + threadIdx.x;
    if (i < n) out[i] = f32_to_bf16(in[i]);
}

// relayout 3x3 weights (cout,cin,3,3) -> bf16 [tap(9)][cout][cin]
__global__ void k_cvt_w3x3(const float* __restrict__ w,
                           unsigned short* __restrict__ out) {
    int i = blockIdx.x * 256 + threadIdx.x;   // 9*128*128
    if (i >= 9 * NC * NC) return;
    int ci = i & 127, co = (i >> 7) & 127, tap = i >> 14;
    out[i] = f32_to_bf16(w[(co * NC + ci) * 9 + tap]);
}

// bilinear 4x upsample (align-corners linspace), S:(B,128,32,32)->bf16
__global__ void k_upsample4_bf16(const float* __restrict__ S,
                                 unsigned short* __restrict__ out) {
    int i = blockIdx.x * 256 + threadIdx.x;
    if (i >= NB * NC * HW) return;
    int x = i & 127, y = (i >> 7) & 127, bc = i >> 14;
    float fy = y * (31.0f / 127.0f), fx = x * (31.0f / 127.0f);
    int y0 = (int)floorf(fy), x0 = (int)floorf(fx);
    int y1 = min(y0 + 1, 31), x1 = min(x0 + 1, 31);
    float wy = fy - (float)y0, wx = fx - (float)x0;
    const float* p = S + (size_t)bc * 1024;
    float v00 = p[y0 * 32 + x0], v01 = p[y0 * 32 + x1];
    float v10 = p[y1 * 32 + x0], v11 = p[y1 * 32 + x1];
    float top = v00 * (1.f - wx) + v01 * wx;
    float bot = v10 * (1.f - wx) + v11 * wx;
    out[i] = f32_to_bf16(top * (1.f - wy) + bot * wy);
}

// ---------------------------------------------------------------------------
// 3x3 conv 128->128, implicit GEMM. LDS tile [dy][x][cin] (cin innermost) so
// each A fragment is two ds_load_b128; weights [tap][cout][cin] so each B
// fragment is two global_load_b128. BN + ReLU6 epilogue, optional per-(b,cin)
// input scale (folds "R + lc*R").
__global__ __launch_bounds__(256) void k_conv3x3_wmma(
    const unsigned short* __restrict__ in_bf,
    const unsigned short* __restrict__ w_bf,   // [tap][cout][cin] bf16
    const float* __restrict__ bn_s, const float* __restrict__ bn_b,
    const float* __restrict__ in_scale,        // [B*128] or nullptr
    float* __restrict__ out_f32, unsigned short* __restrict__ out_bf) {
    __shared__ __attribute__((aligned(16))) unsigned short tile[3 * 18 * 128];
    const int b = blockIdx.z, y = blockIdx.y, x0 = blockIdx.x * 16;
    const int tid = threadIdx.x;
    for (int idx = tid; idx < 3 * 18 * 128; idx += 256) {
        int cin = idx / 54, rem = idx % 54, dy = rem / 18, xx = rem % 18;
        int gy = y - 1 + dy, gx = x0 - 1 + xx;
        float v = 0.f;
        if (gy >= 0 && gy < 128 && gx >= 0 && gx < 128) {
            v = bf16_to_f32(in_bf[(((size_t)b * NC + cin) << 14) + (gy << 7) + gx]);
            if (in_scale) v *= in_scale[b * NC + cin];
        }
        tile[(dy * 18 + xx) * 128 + cin] = f32_to_bf16(v);   // cin innermost
    }
    __syncthreads();
    const int wave = tid >> 5, lane = tid & 31, half = lane >> 4, l16 = lane & 15;
    const int n = wave * 16 + l16;   // cout column for this lane
    v8f acc = {};
    for (int tap = 0; tap < 9; ++tap) {
        int dy = tap / 3, dx = tap % 3;
        const unsigned short* wrow = w_bf + ((size_t)tap * NC + n) * NC;
        if (tap < 8)
            __builtin_prefetch(w_bf + ((size_t)(tap + 1) * NC + n) * NC, 0, 1);
        const unsigned short* arow = tile + (dy * 18 + l16 + dx) * 128;
#pragma unroll
        for (int kc = 0; kc < 4; ++kc) {
            int kb = kc * 32 + half * 8;
            u32x4 alo = *(const u32x4*)(arow + kb);
            u32x4 ahi = *(const u32x4*)(arow + kb + 16);
            u32x4 blo = *(const u32x4*)(wrow + kb);
            u32x4 bhi = *(const u32x4*)(wrow + kb + 16);
            acc = wmma_bf16(make_frag(alo, ahi), make_frag(blo, bhi), acc);
        }
    }
#pragma unroll
    for (int vv = 0; vv < 8; ++vv) {
        int m = vv + 8 * half;
        float val = acc[vv] * bn_s[n] + bn_b[n];
        val = fminf(fmaxf(val, 0.f), 6.f);                    // ReLU6
        size_t oi = (((size_t)b * NC + n) << 14) + (y << 7) + (x0 + m);
        if (out_f32) out_f32[oi] = val;
        else         out_bf[oi] = f32_to_bf16(val);
    }
}

// ---------------------------------------------------------------------------
// 1x1 conv 128->128 GEMM. LDS [m][cin]; A = 2x ds_load_b128, B = 2x b128.
// act: 0 = raw, 1 = BN + ReLU
__global__ __launch_bounds__(256) void k_conv1x1_wmma(
    const unsigned short* __restrict__ in_bf,
    const unsigned short* __restrict__ w_bf,   // [cout][cin] bf16
    const float* __restrict__ bn_s, const float* __restrict__ bn_b,
    int act, float* __restrict__ out) {
    __shared__ __attribute__((aligned(16))) unsigned short ina[16 * 128];
    const int b = blockIdx.y, n0 = blockIdx.x * 16, tid = threadIdx.x;
    for (int idx = tid; idx < 1024; idx += 256) {
        int cin = idx >> 3, mp = idx & 7;
        unsigned w2 = *(const unsigned*)(in_bf + (((size_t)b * NC + cin) << 14) + n0 + 2 * mp);
        ina[(2 * mp) * 128 + cin]     = (unsigned short)(w2 & 0xFFFFu);
        ina[(2 * mp + 1) * 128 + cin] = (unsigned short)(w2 >> 16);
    }
    __syncthreads();
    const int wave = tid >> 5, lane = tid & 31, half = lane >> 4, l16 = lane & 15;
    const int n = wave * 16 + l16;
    const unsigned short* arow = ina + l16 * 128;
    const unsigned short* wrow = w_bf + (size_t)n * NC;
    v8f acc = {};
#pragma unroll
    for (int kc = 0; kc < 4; ++kc) {
        int kb = kc * 32 + half * 8;
        u32x4 alo = *(const u32x4*)(arow + kb);
        u32x4 ahi = *(const u32x4*)(arow + kb + 16);
        u32x4 blo = *(const u32x4*)(wrow + kb);
        u32x4 bhi = *(const u32x4*)(wrow + kb + 16);
        acc = wmma_bf16(make_frag(alo, ahi), make_frag(blo, bhi), acc);
    }
#pragma unroll
    for (int vv = 0; vv < 8; ++vv) {
        int m = vv + 8 * half;
        float val = acc[vv];
        if (act == 1) val = fmaxf(val * bn_s[n] + bn_b[n], 0.f);
        out[(((size_t)b * NC + n) << 14) + n0 + m] = val;
    }
}

// ---------------------------------------------------------------------------
// 1x1 classifier conv to 6 classes (tiny -> VALU)
__global__ void k_cls(const unsigned short* __restrict__ dec_bf,
                      const float* __restrict__ wcls, float* __restrict__ D6) {
    int i = blockIdx.x * 256 + threadIdx.x;   // B*6*HW
    if (i >= NB * 6 * HW) return;
    int n = i & (HW - 1);
    int k = (i >> 14) % 6;
    int b = i / (6 * HW);
    float acc = 0.f;
    for (int c = 0; c < NC; ++c)
        acc += bf16_to_f32(dec_bf[(((size_t)b * NC + c) << 14) + n]) * wcls[k * NC + c];
    D6[i] = acc;
}

// spatial softmax per (b, class) in place
__global__ __launch_bounds__(256) void k_softmax_spatial(float* __restrict__ D6) {
    __shared__ float red[256];
    const int k = blockIdx.x, b = blockIdx.y, tid = threadIdx.x;
    float* p = D6 + ((size_t)b * 6 + k) * HW;
    float m = -3.4e38f;
    for (int n = tid; n < HW; n += 256) m = fmaxf(m, p[n]);
    red[tid] = m; __syncthreads();
    for (int s = 128; s > 0; s >>= 1) { if (tid < s) red[tid] = fmaxf(red[tid], red[tid + s]); __syncthreads(); }
    m = red[0]; __syncthreads();
    float sum = 0.f;
    for (int n = tid; n < HW; n += 256) sum += __expf(p[n] - m);
    red[tid] = sum; __syncthreads();
    for (int s = 128; s > 0; s >>= 1) { if (tid < s) red[tid] += red[tid + s]; __syncthreads(); }
    float inv = 1.f / red[0]; __syncthreads();
    for (int n = tid; n < HW; n += 256) p[n] = __expf(p[n] - m) * inv;
}

// SBC[b,c] = 1 + mean_k sum_n probs[b,k,n] * R[b,c,n]
__global__ __launch_bounds__(256) void k_ctx_scale(const float* __restrict__ D6,
                                                   const float* __restrict__ R,
                                                   float* __restrict__ SBC) {
    __shared__ float red[256];
    const int c = blockIdx.x, b = blockIdx.y, tid = threadIdx.x;
    const float* r = R + (((size_t)b * NC + c) << 14);
    const float* p = D6 + (size_t)b * 6 * HW;
    float acc = 0.f;
    for (int n = tid; n < HW; n += 256) {
        float ps = p[n] + p[HW + n] + p[2 * HW + n] + p[3 * HW + n] + p[4 * HW + n] + p[5 * HW + n];
        acc += r[n] * ps;
    }
    red[tid] = acc; __syncthreads();
    for (int s = 128; s > 0; s >>= 1) { if (tid < s) red[tid] += red[tid + s]; __syncthreads(); }
    if (tid == 0) SBC[b * NC + c] = 1.f + red[0] * (1.f / 6.f);
}

// ---------------------------------------------------------------------------
// Windowed attention: one wave per (window, head). RoPE+scale folded into bf16
// LDS staging; QK^T (K=8 padded->32) and A*V (N=8 padded->16) in WMMA bf16,
// all fragments as aligned b128 LDS loads.
__global__ __launch_bounds__(32) void k_attn(
    const float* __restrict__ Q, const float* __restrict__ Km,
    const float* __restrict__ V, const float* __restrict__ bias_table,
    const float* __restrict__ rope, float* __restrict__ OUT) {
    __shared__ __attribute__((aligned(16))) unsigned short qbf[64 * 8];
    __shared__ __attribute__((aligned(16))) unsigned short kbf[64 * 8];
    __shared__ __attribute__((aligned(16))) unsigned short vtb[8 * 64];   // [dc][pos]
    __shared__ __attribute__((aligned(16))) unsigned short pbf[64 * 64];  // probs bf16
    __shared__ float dots[64][64];
    const int b = blockIdx.z, h = blockIdx.y;
    const int wy = blockIdx.x >> 4, wx = blockIdx.x & 15;
    const int lane = threadIdx.x, half = lane >> 4, l16 = lane & 15;
    const float scale = 0.35355339059327373f;   // 8^-0.5
    for (int r = lane; r < 64; r += 32) {
        int ty = r >> 3, tx = r & 7;
        int y = wy * 8 + ty, x = wx * 8 + tx;
        size_t base = (((size_t)b * NC + h * 8) << 14) + (y << 7) + x;
#pragma unroll
        for (int m2 = 0; m2 < 4; ++m2) {
            float fx = rope[h * 4 + m2];
            float fy = rope[64 + h * 4 + m2];
            float ang = tx * fx + ty * fy;
            float cs = __cosf(ang), sn = __sinf(ang);
            float qa = Q[base + ((size_t)(2 * m2) << 14)];
            float qb = Q[base + ((size_t)(2 * m2 + 1) << 14)];
            qbf[r * 8 + 2 * m2]     = f32_to_bf16((qa * cs - qb * sn) * scale);
            qbf[r * 8 + 2 * m2 + 1] = f32_to_bf16((qa * sn + qb * cs) * scale);
            float ka = Km[base + ((size_t)(2 * m2) << 14)];
            float kb = Km[base + ((size_t)(2 * m2 + 1) << 14)];
            kbf[r * 8 + 2 * m2]     = f32_to_bf16(ka * cs - kb * sn);
            kbf[r * 8 + 2 * m2 + 1] = f32_to_bf16(ka * sn + kb * cs);
        }
#pragma unroll
        for (int dc = 0; dc < 8; ++dc)
            vtb[dc * 64 + r] = f32_to_bf16(V[base + ((size_t)dc << 14)]);
    }
    __syncthreads();
    const u32x4 z4 = {};
    // dots = q . k^T  (4x4 tiles; K=8, only half==0 lanes carry data)
    for (int mt = 0; mt < 4; ++mt)
        for (int nt = 0; nt < 4; ++nt) {
            u32x4 qlo = z4, klo = z4;
            if (half == 0) {
                qlo = *(const u32x4*)(qbf + (mt * 16 + l16) * 8);
                klo = *(const u32x4*)(kbf + (nt * 16 + l16) * 8);
            }
            v8f acc = {};
            acc = wmma_bf16(make_frag(qlo, z4), make_frag(klo, z4), acc);
#pragma unroll
            for (int vv = 0; vv < 8; ++vv)
                dots[mt * 16 + vv + 8 * half][nt * 16 + l16] = acc[vv];
        }
    __syncthreads();
    // relative bias + row softmax (2 rows per lane) -> bf16 probabilities
    for (int r = lane; r < 64; r += 32) {
        int ty = r >> 3, tx = r & 7;
        float mx = -3.4e38f;
        for (int c2 = 0; c2 < 64; ++c2) {
            int cy = c2 >> 3, cx = c2 & 7;
            int ridx = (ty - cy + 7) * 15 + (tx - cx + 7);
            float v2 = dots[r][c2] + bias_table[ridx * 16 + h];
            dots[r][c2] = v2;
            mx = fmaxf(mx, v2);
        }
        float sum = 0.f;
        for (int c2 = 0; c2 < 64; ++c2) { float e2 = __expf(dots[r][c2] - mx); dots[r][c2] = e2; sum += e2; }
        float inv = 1.f / sum;
        for (int c2 = 0; c2 < 64; ++c2) pbf[r * 64 + c2] = f32_to_bf16(dots[r][c2] * inv);
    }
    __syncthreads();
    // out = attn @ v : M=64 (4 tiles) x N=8 (padded 16), K=64 (2 chunks)
    for (int mt = 0; mt < 4; ++mt) {
        const unsigned short* prow = pbf + (mt * 16 + l16) * 64;
        v8f acc = {};
#pragma unroll
        for (int kc = 0; kc < 2; ++kc) {
            int kb = kc * 32 + half * 8;
            u32x4 alo = *(const u32x4*)(prow + kb);
            u32x4 ahi = *(const u32x4*)(prow + kb + 16);
            u32x4 blo = z4, bhi = z4;
            if (l16 < 8) {
                const unsigned short* vrow = vtb + l16 * 64;
                blo = *(const u32x4*)(vrow + kb);
                bhi = *(const u32x4*)(vrow + kb + 16);
            }
            acc = wmma_bf16(make_frag(alo, ahi), make_frag(blo, bhi), acc);
        }
        if (l16 < 8) {
#pragma unroll
            for (int vv = 0; vv < 8; ++vv) {
                int r = mt * 16 + vv + 8 * half;
                int ty = r >> 3, tx = r & 7;
                int y = wy * 8 + ty, x = wx * 8 + tx;
                OUT[(((size_t)b * NC + h * 8 + l16) << 14) + (y << 7) + x] = acc[vv];
            }
        }
    }
}

// ---------------------------------------------------------------------------
// head-pool of attn_map: mean over d within each head
__global__ void k_hpool(const float* __restrict__ in, float* __restrict__ out) {
    int i = blockIdx.x * 256 + threadIdx.x;   // B*16*HW
    if (i >= NB * 16 * HW) return;
    int n = i & (HW - 1), hd = (i >> 14) & 15, b = i >> 18;
    float acc = 0.f;
    for (int d = 0; d < 8; ++d) acc += in[(((size_t)b * NC + hd * 8 + d) << 14) + n];
    out[i] = acc * 0.125f;
}

// head-pool of the 8-tap line average of v (axis 0: vertical, 1: horizontal)
// reflect-pad by 1 (idx 128 -> 126), zero outside.
__global__ void k_hpool_line(const float* __restrict__ V, float* __restrict__ out, int axis) {
    int i = blockIdx.x * 256 + threadIdx.x;
    if (i >= NB * 16 * HW) return;
    int x = i & 127, y = (i >> 7) & 127, hd = (i >> 14) & 15, b = i >> 18;
    float acc = 0.f;
    for (int d = 0; d < 8; ++d) {
        const float* p = V + (((size_t)b * NC + hd * 8 + d) << 14);
#pragma unroll
        for (int t = 0; t < 8; ++t) {
            int idx = ((axis == 0) ? y : x) - 3 + t;
            if (idx < 0 || idx > 128) continue;
            if (idx == 128) idx = 126;
            acc += (axis == 0) ? p[(idx << 7) + x] : p[(y << 7) + idx];
        }
    }
    out[i] = acc * (1.f / 64.f);
}

// small 3x3 conv for the clc gates (16 out ch). act: 0=leaky(0.1), 1=sigmoid
__global__ void k_conv3x3_small(const float* __restrict__ inA, const float* __restrict__ inB,
                                const float* __restrict__ w, int cinA, int cinB,
                                int act, float* __restrict__ out) {
    int i = blockIdx.x * 256 + threadIdx.x;   // B*16*HW
    if (i >= NB * 16 * HW) return;
    int x = i & 127, y = (i >> 7) & 127, co = (i >> 14) & 15, b = i >> 18;
    int cint = cinA + cinB;
    float acc = 0.f;
    for (int c = 0; c < cint; ++c) {
        const float* p = (c < cinA) ? (inA + (((size_t)b * cinA + c) << 14))
                                    : (inB + (((size_t)b * cinB + (c - cinA)) << 14));
        for (int dy = 0; dy < 3; ++dy) {
            int yy = y - 1 + dy; if (yy < 0 || yy >= 128) continue;
            for (int dx = 0; dx < 3; ++dx) {
                int xx = x - 1 + dx; if (xx < 0 || xx >= 128) continue;
                acc += p[(yy << 7) + xx] * w[((co * cint + c) * 3 + dy) * 3 + dx];
            }
        }
    }
    out[i] = (act == 0) ? ((acc > 0.f) ? acc : 0.1f * acc)
                        : 1.f / (1.f + __expf(-acc));
}

// comb = attn + rep(g1)*lx + rep(g2)*ly, lx/ly recomputed on the fly from v
__global__ void k_comb(const float* __restrict__ ATT, const float* __restrict__ g1,
                       const float* __restrict__ g2, const float* __restrict__ V,
                       float* __restrict__ out) {
    int i = blockIdx.x * 256 + threadIdx.x;   // B*128*HW
    if (i >= (NB << 21)) return;
    int x = i & 127, y = (i >> 7) & 127, c = (i >> 14) & 127, b = i >> 21;
    const float* p = V + (((size_t)b * NC + c) << 14);
    float lx = 0.f, ly = 0.f;
#pragma unroll
    for (int t = 0; t < 8; ++t) {
        int yy = y - 3 + t;
        if (yy >= 0 && yy <= 128) { if (yy == 128) yy = 126; lx += p[(yy << 7) + x]; }
        int xx = x - 3 + t;
        if (xx >= 0 && xx <= 128) { if (xx == 128) xx = 126; ly += p[(y << 7) + xx]; }
    }
    lx *= 0.125f; ly *= 0.125f;
    int hd = c >> 3;
    size_t gi = (((size_t)b * 16 + hd) << 14) + (y << 7) + x;
    out[i] = ATT[i] + g1[gi] * lx + g2[gi] * ly;
}

// depthwise 8x8 conv with reflect(+1)/zero(+3) padding + BN -> bf16
__global__ void k_dw(const float* __restrict__ in, const float* __restrict__ w,
                     const float* __restrict__ bn_s, const float* __restrict__ bn_b,
                     unsigned short* __restrict__ out) {
    int i = blockIdx.x * 256 + threadIdx.x;
    if (i >= (NB << 21)) return;
    int x = i & 127, y = (i >> 7) & 127, c = (i >> 14) & 127, b = i >> 21;
    const float* p = in + (((size_t)b * NC + c) << 14);
    const float* wk = w + c * 64;
    float acc = 0.f;
    for (int ky = 0; ky < 8; ++ky) {
        int yy = y - 3 + ky;
        if (yy < 0 || yy > 128) continue;
        int ys = (yy == 128) ? 126 : yy;
        for (int kx = 0; kx < 8; ++kx) {
            int xx = x - 3 + kx;
            if (xx < 0 || xx > 128) continue;
            int xs = (xx == 128) ? 126 : xx;
            acc += p[(ys << 7) + xs] * wk[ky * 8 + kx];
        }
    }
    out[i] = f32_to_bf16(acc * bn_s[c] + bn_b[c]);
}

// ---------------------------------------------------------------------------
extern "C" void kernel_launch(void* const* d_in, const int* in_sizes, int n_in,
                              void* d_out, int out_size, void* d_ws, size_t ws_size,
                              hipStream_t stream) {
    (void)in_sizes; (void)n_in; (void)out_size; (void)ws_size;
    const float* R          = (const float*)d_in[0];
    const float* S          = (const float*)d_in[1];
    const float* dec_conv_w = (const float*)d_in[2];
    const float* dec_bn_s   = (const float*)d_in[3];
    const float* dec_bn_b   = (const float*)d_in[4];
    const float* dec_cls_w  = (const float*)d_in[5];
    const float* post_w     = (const float*)d_in[6];
    const float* post_bn_s  = (const float*)d_in[7];
    const float* post_bn_b  = (const float*)d_in[8];
    const float* q_w        = (const float*)d_in[9];
    const float* q_bn_s     = (const float*)d_in[10];
    const float* q_bn_b     = (const float*)d_in[11];
    const float* k_w        = (const float*)d_in[12];
    const float* k_bn_s     = (const float*)d_in[13];
    const float* k_bn_b     = (const float*)d_in[14];
    const float* v_w        = (const float*)d_in[15];
    const float* v_bn_s     = (const float*)d_in[16];
    const float* v_bn_b     = (const float*)d_in[17];
    const float* clc1_w1    = (const float*)d_in[18];
    const float* clc1_w2    = (const float*)d_in[19];
    const float* clc2_w1    = (const float*)d_in[20];
    const float* clc2_w2    = (const float*)d_in[21];
    const float* proj_dw_w  = (const float*)d_in[22];
    const float* proj_bn_s  = (const float*)d_in[23];
    const float* proj_bn_b  = (const float*)d_in[24];
    const float* proj_pw_w  = (const float*)d_in[25];
    const float* rel_bias   = (const float*)d_in[26];
    const float* rope_freqs = (const float*)d_in[27];

    const unsigned long long MB = 1024ull * 1024ull;
    char* ws = (char*)d_ws;
    unsigned short* Rbf    = (unsigned short*)(ws + 0);        // 16MB
    unsigned short* S4bf   = (unsigned short*)(ws + 16 * MB);  // 16MB
    unsigned short* TMPBF  = (unsigned short*)(ws + 32 * MB);  // dec-out -> cR -> dw-out (bf16)
    float*          D6     = (float*)(ws + 48 * MB);           // 1.5MB
    float*          SBC    = (float*)(ws + 50 * MB);           // 2KB
    unsigned short* decwbf = (unsigned short*)(ws + 51 * MB);  // [tap][co][ci]
    unsigned short* postwbf= (unsigned short*)(ws + 52 * MB);  // [tap][co][ci]
    unsigned short* qwbf   = (unsigned short*)(ws + 53 * MB);
    unsigned short* kwbf   = qwbf + 16384;
    unsigned short* vwbf   = kwbf + 16384;
    unsigned short* pwwbf  = vwbf + 16384;
    float*          Qb     = (float*)(ws + 54 * MB);           // 32MB (later: comb)
    float*          Kb     = (float*)(ws + 86 * MB);
    float*          Vb     = (float*)(ws + 118 * MB);
    float*          ATT    = (float*)(ws + 150 * MB);
    // reuse regions (stream-ordered, consumers of originals already done):
    float* hpA  = (float*)(ws + 0);
    float* hpLx = (float*)(ws + 4 * MB);
    float* hpLy = (float*)(ws + 8 * MB);
    float* ytmp = (float*)(ws + 12 * MB);
    float* g1   = (float*)(ws + 16 * MB);
    float* g2   = (float*)(ws + 20 * MB);
    float* COMB = Qb;

    const int NTEN = NB * NC * HW;   // 8388608

    // 0) bf16 conversions / weight relayouts + upsample
    k_cvt_bf16<<<(NTEN + 255) / 256, 256, 0, stream>>>(R, Rbf, NTEN);
    k_cvt_w3x3<<<(9 * NC * NC + 255) / 256, 256, 0, stream>>>(dec_conv_w, decwbf);
    k_cvt_w3x3<<<(9 * NC * NC + 255) / 256, 256, 0, stream>>>(post_w, postwbf);
    k_cvt_bf16<<<64, 256, 0, stream>>>(q_w, qwbf, 16384);
    k_cvt_bf16<<<64, 256, 0, stream>>>(k_w, kwbf, 16384);
    k_cvt_bf16<<<64, 256, 0, stream>>>(v_w, vwbf, 16384);
    k_cvt_bf16<<<64, 256, 0, stream>>>(proj_pw_w, pwwbf, 16384);
    k_upsample4_bf16<<<NTEN / 256, 256, 0, stream>>>(S, S4bf);

    // 1) decoder conv3x3 + BN + ReLU6 -> bf16
    k_conv3x3_wmma<<<dim3(8, 128, 4), 256, 0, stream>>>(
        Rbf, decwbf, dec_bn_s, dec_bn_b, nullptr, nullptr, TMPBF);
    // 2) classifier -> spatial softmax -> ctx -> scale (1+lc)
    k_cls<<<(NB * 6 * HW + 255) / 256, 256, 0, stream>>>(TMPBF, dec_cls_w, D6);
    k_softmax_spatial<<<dim3(6, 4), 256, 0, stream>>>(D6);
    k_ctx_scale<<<dim3(128, 4), 256, 0, stream>>>(D6, R, SBC);
    // 3) post conv3x3 on (1+lc)*R (scale folded into LDS fill) -> cR bf16
    k_conv3x3_wmma<<<dim3(8, 128, 4), 256, 0, stream>>>(
        Rbf, postwbf, post_bn_s, post_bn_b, SBC, nullptr, TMPBF);
    // 4) q/k/v 1x1 GEMMs with BN+ReLU
    k_conv1x1_wmma<<<dim3(1024, 4), 256, 0, stream>>>(Rbf,   qwbf, q_bn_s, q_bn_b, 1, Qb);
    k_conv1x1_wmma<<<dim3(1024, 4), 256, 0, stream>>>(TMPBF, kwbf, k_bn_s, k_bn_b, 1, Kb);
    k_conv1x1_wmma<<<dim3(1024, 4), 256, 0, stream>>>(S4bf,  vwbf, v_bn_s, v_bn_b, 1, Vb);
    // 5) windowed attention (RoPE + rel-bias softmax), one wave per (window, head)
    k_attn<<<dim3(256, 16, 4), 32, 0, stream>>>(Qb, Kb, Vb, rel_bias, rope_freqs, ATT);
    // 6) head pools + clc gates
    k_hpool<<<NB * 16 * HW / 256, 256, 0, stream>>>(ATT, hpA);
    k_hpool_line<<<NB * 16 * HW / 256, 256, 0, stream>>>(Vb, hpLx, 0);
    k_hpool_line<<<NB * 16 * HW / 256, 256, 0, stream>>>(Vb, hpLy, 1);
    k_conv3x3_small<<<NB * 16 * HW / 256, 256, 0, stream>>>(hpA, hpLx, clc1_w1, 16, 16, 0, ytmp);
    k_conv3x3_small<<<NB * 16 * HW / 256, 256, 0, stream>>>(ytmp, nullptr, clc1_w2, 16, 0, 1, g1);
    k_conv3x3_small<<<NB * 16 * HW / 256, 256, 0, stream>>>(hpA, hpLy, clc2_w1, 16, 16, 0, ytmp);
    k_conv3x3_small<<<NB * 16 * HW / 256, 256, 0, stream>>>(ytmp, nullptr, clc2_w2, 16, 0, 1, g2);
    // 7) combine + depthwise 8x8 + BN -> bf16
    k_comb<<<NTEN / 256, 256, 0, stream>>>(ATT, g1, g2, Vb, COMB);
    k_dw<<<NTEN / 256, 256, 0, stream>>>(COMB, proj_dw_w, proj_bn_s, proj_bn_b, TMPBF);
    // 8) final pointwise 1x1 GEMM -> d_out (f32)
    k_conv1x1_wmma<<<dim3(1024, 4), 256, 0, stream>>>(TMPBF, pwwbf, nullptr, nullptr, 0,
                                                      (float*)d_out);
}
